// MultiHeadedSelfAttention_85289460564470
// MI455X (gfx1250) — matrix operand order repaired and checked by reading, not compiled, not run on hardware
//
#include <hip/hip_runtime.h>

typedef _Float16 h8_t  __attribute__((ext_vector_type(8)));
typedef _Float16 h16_t __attribute__((ext_vector_type(16)));
typedef float    f8_t  __attribute__((ext_vector_type(8)));

#define BATCH 2
#define SEQ   2048
#define DIM   1024
#define HEADS 16
#define DHEAD 64
#define MTOT  (BATCH * SEQ)   // 4096
#define NQKV  (3 * DIM)       // 3072

static __device__ __forceinline__ h16_t combine16(h8_t lo, h8_t hi) {
    return __builtin_shufflevector(lo, hi, 0, 1, 2, 3, 4, 5, 6, 7, 8, 9, 10, 11, 12, 13, 14, 15);
}

static __device__ __forceinline__ f8_t wmma_f16(h16_t a, h16_t b, f8_t c) {
    // 8 args: (neg_a, A, neg_b, B, c_mod, C, reuse_a, reuse_b)
    return __builtin_amdgcn_wmma_f32_16x16x32_f16(false, a, false, b, (short)0, c, false, false);
}

// Async DMA: global (16B per lane) -> LDS, tracked by ASYNCcnt.
// LDS destination = low 32 bits of the generic __shared__ pointer (LDS offset).
static __device__ __forceinline__ void async_copy_b128(_Float16* lds_dst, const _Float16* gsrc) {
    unsigned lds_off = (unsigned)(unsigned long long)lds_dst;
    unsigned long long gaddr = (unsigned long long)gsrc;
    asm volatile("global_load_async_to_lds_b128 %0, %1, off"
                 :: "v"(lds_off), "v"(gaddr) : "memory");
}

static __device__ __forceinline__ void wait_async0() {
    asm volatile("s_wait_asynccnt 0x0" ::: "memory");
}

// ---------------------------------------------------------------- conversions
__global__ void cvt_f32_to_f16_kernel(const float* __restrict__ src,
                                      _Float16* __restrict__ dst, int n) {
    int i = blockIdx.x * blockDim.x + threadIdx.x;
    if (i < n) dst[i] = (_Float16)src[i];
}

// src: [K][N] f32 row-major  ->  dst: [N][K] f16
__global__ void transpose_cvt_kernel(const float* __restrict__ src,
                                     _Float16* __restrict__ dst, int K, int N) {
    int i = blockIdx.x * blockDim.x + threadIdx.x;
    if (i < K * N) {
        int k = i / N, n = i % N;
        dst[(size_t)n * K + k] = (_Float16)src[i];
    }
}

// ---------------------------------------------------------------- QKV GEMM
// C[M=4096, N=3072] = xh[M, 1024] * wqkvT[N, 1024]^T ; scatter into q/k/v
__global__ __launch_bounds__(256) void qkv_gemm_kernel(
    const _Float16* __restrict__ xh, const _Float16* __restrict__ wT,
    _Float16* __restrict__ qd, _Float16* __restrict__ kd, _Float16* __restrict__ vd) {
    __shared__ __align__(16) _Float16 As[2][128][40];
    __shared__ __align__(16) _Float16 Bs[2][128][40];

    const int tid  = threadIdx.x;
    const int lane = tid & 31;
    const int wave = tid >> 5;
    const int wr   = wave >> 2;     // 0..1 : 64-row strip
    const int wc   = wave & 3;      // 0..3 : 32-col strip
    const int r16  = lane & 15;
    const int hi16 = lane >> 4;
    const int m0   = blockIdx.y * 128;
    const int n0   = blockIdx.x * 128;

    f8_t acc[4][2];
#pragma unroll
    for (int i = 0; i < 4; ++i)
#pragma unroll
        for (int nn = 0; nn < 2; ++nn) acc[i][nn] = (f8_t)0.0f;

    // each thread owns 2 chunks of 8 halves for A and for B per K-tile
    const int row0 = tid >> 2;             // chunks 0..255
    const int kc0  = (tid & 3) * 8;
    const int row1 = (256 + tid) >> 2;     // chunks 256..511
    const int kc1  = ((256 + tid) & 3) * 8;

    // prologue: async-stage tile 0 into buffer 0
    async_copy_b128(&As[0][row0][kc0], &xh[(size_t)(m0 + row0) * DIM + kc0]);
    async_copy_b128(&Bs[0][row0][kc0], &wT[(size_t)(n0 + row0) * DIM + kc0]);
    async_copy_b128(&As[0][row1][kc1], &xh[(size_t)(m0 + row1) * DIM + kc1]);
    async_copy_b128(&Bs[0][row1][kc1], &wT[(size_t)(n0 + row1) * DIM + kc1]);

    int buf = 0;
    for (int k0 = 0; k0 < DIM; k0 += 32, buf ^= 1) {
        wait_async0();          // this wave's tile-k DMA done
        __syncthreads();        // everyone's tile-k in LDS; prev compute finished
        if (k0 + 32 < DIM) {    // pipeline tile k+1 into the other buffer
            int kn = k0 + 32;
            async_copy_b128(&As[buf ^ 1][row0][kc0], &xh[(size_t)(m0 + row0) * DIM + kn + kc0]);
            async_copy_b128(&Bs[buf ^ 1][row0][kc0], &wT[(size_t)(n0 + row0) * DIM + kn + kc0]);
            async_copy_b128(&As[buf ^ 1][row1][kc1], &xh[(size_t)(m0 + row1) * DIM + kn + kc1]);
            async_copy_b128(&Bs[buf ^ 1][row1][kc1], &wT[(size_t)(n0 + row1) * DIM + kn + kc1]);
        }

        h16_t af[4], bf[2];
#pragma unroll
        for (int i = 0; i < 4; ++i) {
            int row = wr * 64 + i * 16 + r16;
            int kb  = hi16 * 8;
            af[i] = combine16(*(const h8_t*)&As[buf][row][kb], *(const h8_t*)&As[buf][row][kb + 16]);
        }
#pragma unroll
        for (int nn = 0; nn < 2; ++nn) {
            int col = wc * 32 + nn * 16 + r16;
            int kb2 = hi16 * 8;
            bf[nn] = combine16(*(const h8_t*)&Bs[buf][col][kb2], *(const h8_t*)&Bs[buf][col][kb2 + 16]);
        }
#pragma unroll
        for (int i = 0; i < 4; ++i)
#pragma unroll
            for (int nn = 0; nn < 2; ++nn)
                acc[i][nn] = wmma_f16(af[i], bf[nn], acc[i][nn]);
    }

    // epilogue: column c decomposes as c = d*48 + trip*16 + head
#pragma unroll
    for (int i = 0; i < 4; ++i) {
#pragma unroll
        for (int nn = 0; nn < 2; ++nn) {
            int c    = n0 + wc * 32 + nn * 16 + r16;
            int d    = c / 48;
            int trip = (c >> 4) % 3;
            int head = c & 15;
#pragma unroll
            for (int j = 0; j < 8; ++j) {
                int m  = m0 + wr * 64 + i * 16 + j + hi16 * 8;
                int bi = m >> 11;
                int t  = m & (SEQ - 1);
                size_t off = ((size_t)(bi * HEADS + head) * SEQ + t) * DHEAD + d;
                float v = acc[i][nn][j];
                if (trip == 0)      qd[off] = (_Float16)(v * 0.125f);  // fold 1/sqrt(dh)
                else if (trip == 1) kd[off] = (_Float16)v;
                else                vd[off] = (_Float16)v;
            }
        }
    }
}

// ---------------------------------------------------------------- attention
__global__ __launch_bounds__(256) void attn_kernel(
    const _Float16* __restrict__ qg, const _Float16* __restrict__ kg,
    const _Float16* __restrict__ vg, _Float16* __restrict__ ctx) {
    __shared__ __align__(16) _Float16 Ks[64][72];       // [key][dh]
    __shared__ __align__(16) _Float16 Vt[64][72];       // [dh][key]
    __shared__ __align__(16) _Float16 Ps[8][16][72];    // per-wave P tile

    const int tid  = threadIdx.x;
    const int lane = tid & 31;
    const int wave = tid >> 5;
    const int r16  = lane & 15;
    const int hi16 = lane >> 4;
    const int bh   = blockIdx.y;
    const int bi   = bh >> 4;
    const int head = bh & 15;
    const size_t base = (size_t)(bi * HEADS + head) * SEQ * DHEAD;
    const int q0 = blockIdx.x * 128 + wave * 16;        // 16 queries per wave

    // Q fragments resident in registers for the whole kernel (pre-scaled)
    h16_t qf[2];
#pragma unroll
    for (int ks = 0; ks < 2; ++ks) {
        const _Float16* qp = qg + base + (size_t)(q0 + r16) * DHEAD + ks * 32 + hi16 * 8;
        qf[ks] = combine16(*(const h8_t*)qp, *(const h8_t*)(qp + 16));
    }

    f8_t oacc[4];
#pragma unroll
    for (int nt = 0; nt < 4; ++nt) oacc[nt] = (f8_t)0.0f;
    float mstate[8], lstate[8];
#pragma unroll
    for (int j = 0; j < 8; ++j) { mstate[j] = -1e30f; lstate[j] = 0.0f; }

    const int srow = tid >> 2;           // staging coords
    const int skc  = (tid & 3) * 8;

    for (int ktile = 0; ktile < SEQ / 64; ++ktile) {
        __syncthreads();                 // prev iteration's readers done
        {
            const size_t goff = base + (size_t)(ktile * 64 + srow) * DHEAD + skc;
            // K tile: async DMA straight into LDS (row-major, matches B-frag reads)
            async_copy_b128(&Ks[srow][skc], &kg[goff]);
            // V tile: needs transpose -> synchronous path through VGPRs
            h8_t v8 = *(const h8_t*)&vg[goff];
#pragma unroll
            for (int e = 0; e < 8; ++e) Vt[skc + e][srow] = v8[e];
            if (ktile + 1 < SEQ / 64)
                __builtin_prefetch(&vg[goff + 64 * DHEAD], 0, 1);
        }
        wait_async0();
        __syncthreads();

        // S = Q * K^T   (16 x 64, per wave)
        f8_t s[4];
#pragma unroll
        for (int nt = 0; nt < 4; ++nt) {
            s[nt] = (f8_t)0.0f;
#pragma unroll
            for (int ks = 0; ks < 2; ++ks) {
                int key = nt * 16 + r16;
                int kb  = ks * 32 + hi16 * 8;
                h16_t bf = combine16(*(const h8_t*)&Ks[key][kb],
                                     *(const h8_t*)&Ks[key][kb + 16]);
                s[nt] = wmma_f16(qf[ks], bf, s[nt]);
            }
        }

        // online softmax (rows are spread over 16 lanes per half-wave)
        float mnew[8], alpha[8];
#pragma unroll
        for (int j = 0; j < 8; ++j) {
            float mx = fmaxf(fmaxf(s[0][j], s[1][j]), fmaxf(s[2][j], s[3][j]));
#pragma unroll
            for (int msk = 1; msk < 16; msk <<= 1)
                mx = fmaxf(mx, __shfl_xor(mx, msk, 32));
            mnew[j]  = fmaxf(mstate[j], mx);
            alpha[j] = __expf(mstate[j] - mnew[j]);
        }
        float p[4][8];
#pragma unroll
        for (int nt = 0; nt < 4; ++nt)
#pragma unroll
            for (int j = 0; j < 8; ++j) p[nt][j] = __expf(s[nt][j] - mnew[j]);
#pragma unroll
        for (int j = 0; j < 8; ++j) {
            float sum = p[0][j] + p[1][j] + p[2][j] + p[3][j];
#pragma unroll
            for (int msk = 1; msk < 16; msk <<= 1)
                sum += __shfl_xor(sum, msk, 32);
            lstate[j] = lstate[j] * alpha[j] + sum;
            mstate[j] = mnew[j];
        }
#pragma unroll
        for (int nt = 0; nt < 4; ++nt)
#pragma unroll
            for (int j = 0; j < 8; ++j) oacc[nt][j] *= alpha[j];

        // C-layout -> A-fragment layout via per-wave LDS (same-wave DS order)
#pragma unroll
        for (int nt = 0; nt < 4; ++nt)
#pragma unroll
            for (int j = 0; j < 8; ++j)
                Ps[wave][j + hi16 * 8][nt * 16 + r16] = (_Float16)p[nt][j];

        h16_t pf[2];
#pragma unroll
        for (int ks = 0; ks < 2; ++ks) {
            int col = ks * 32 + hi16 * 8;
            pf[ks] = combine16(*(const h8_t*)&Ps[wave][r16][col],
                               *(const h8_t*)&Ps[wave][r16][col + 16]);
        }

        // O += P * V
#pragma unroll
        for (int nt = 0; nt < 4; ++nt)
#pragma unroll
            for (int ks = 0; ks < 2; ++ks) {
                int dh = nt * 16 + r16;
                int kb = ks * 32 + hi16 * 8;
                h16_t bf = combine16(*(const h8_t*)&Vt[dh][kb],
                                     *(const h8_t*)&Vt[dh][kb + 16]);
                oacc[nt] = wmma_f16(pf[ks], bf, oacc[nt]);
            }
    }

    // normalize + write context in [b, t, head*dh + d] layout for the out-proj
#pragma unroll
    for (int nt = 0; nt < 4; ++nt)
#pragma unroll
        for (int j = 0; j < 8; ++j) {
            int t = q0 + j + hi16 * 8;
            int d = nt * 16 + r16;
            float v = oacc[nt][j] / lstate[j];
            ctx[((size_t)bi * SEQ + t) * DIM + head * DHEAD + d] = (_Float16)v;
        }
}

// ---------------------------------------------------------------- output GEMM
__global__ __launch_bounds__(256) void out_gemm_kernel(
    const _Float16* __restrict__ ah, const _Float16* __restrict__ wT,
    float* __restrict__ out) {
    __shared__ __align__(16) _Float16 As[2][128][40];
    __shared__ __align__(16) _Float16 Bs[2][128][40];

    const int tid  = threadIdx.x;
    const int lane = tid & 31;
    const int wave = tid >> 5;
    const int wr   = wave >> 2;
    const int wc   = wave & 3;
    const int r16  = lane & 15;
    const int hi16 = lane >> 4;
    const int m0   = blockIdx.y * 128;
    const int n0   = blockIdx.x * 128;

    f8_t acc[4][2];
#pragma unroll
    for (int i = 0; i < 4; ++i)
#pragma unroll
        for (int nn = 0; nn < 2; ++nn) acc[i][nn] = (f8_t)0.0f;

    const int row0 = tid >> 2;
    const int kc0  = (tid & 3) * 8;
    const int row1 = (256 + tid) >> 2;
    const int kc1  = ((256 + tid) & 3) * 8;

    async_copy_b128(&As[0][row0][kc0], &ah[(size_t)(m0 + row0) * DIM + kc0]);
    async_copy_b128(&Bs[0][row0][kc0], &wT[(size_t)(n0 + row0) * DIM + kc0]);
    async_copy_b128(&As[0][row1][kc1], &ah[(size_t)(m0 + row1) * DIM + kc1]);
    async_copy_b128(&Bs[0][row1][kc1], &wT[(size_t)(n0 + row1) * DIM + kc1]);

    int buf = 0;
    for (int k0 = 0; k0 < DIM; k0 += 32, buf ^= 1) {
        wait_async0();
        __syncthreads();
        if (k0 + 32 < DIM) {
            int kn = k0 + 32;
            async_copy_b128(&As[buf ^ 1][row0][kc0], &ah[(size_t)(m0 + row0) * DIM + kn + kc0]);
            async_copy_b128(&Bs[buf ^ 1][row0][kc0], &wT[(size_t)(n0 + row0) * DIM + kn + kc0]);
            async_copy_b128(&As[buf ^ 1][row1][kc1], &ah[(size_t)(m0 + row1) * DIM + kn + kc1]);
            async_copy_b128(&Bs[buf ^ 1][row1][kc1], &wT[(size_t)(n0 + row1) * DIM + kn + kc1]);
        }

        h16_t af[4], bf[2];
#pragma unroll
        for (int i = 0; i < 4; ++i) {
            int row = wr * 64 + i * 16 + r16;
            int kb  = hi16 * 8;
            af[i] = combine16(*(const h8_t*)&As[buf][row][kb], *(const h8_t*)&As[buf][row][kb + 16]);
        }
#pragma unroll
        for (int nn = 0; nn < 2; ++nn) {
            int col = wc * 32 + nn * 16 + r16;
            int kb2 = hi16 * 8;
            bf[nn] = combine16(*(const h8_t*)&Bs[buf][col][kb2], *(const h8_t*)&Bs[buf][col][kb2 + 16]);
        }
#pragma unroll
        for (int i = 0; i < 4; ++i)
#pragma unroll
            for (int nn = 0; nn < 2; ++nn)
                acc[i][nn] = wmma_f16(af[i], bf[nn], acc[i][nn]);
    }

#pragma unroll
    for (int i = 0; i < 4; ++i)
#pragma unroll
        for (int nn = 0; nn < 2; ++nn) {
            int c = n0 + wc * 32 + nn * 16 + r16;
#pragma unroll
            for (int j = 0; j < 8; ++j) {
                int m = m0 + wr * 64 + i * 16 + j + hi16 * 8;
                out[(size_t)m * DIM + c] = acc[i][nn][j];
            }
        }
}

// ---------------------------------------------------------------- launcher
extern "C" void kernel_launch(void* const* d_in, const int* in_sizes, int n_in,
                              void* d_out, int out_size, void* d_ws, size_t ws_size,
                              hipStream_t stream) {
    const float* x     = (const float*)d_in[0];
    const float* w_qkv = (const float*)d_in[1];
    const float* w_out = (const float*)d_in[2];
    float* out = (float*)d_out;

    char* ws = (char*)d_ws;
    _Float16* xh    = (_Float16*)ws; ws += (size_t)MTOT * DIM * 2;
    _Float16* wqkvT = (_Float16*)ws; ws += (size_t)NQKV * DIM * 2;
    _Float16* woutT = (_Float16*)ws; ws += (size_t)DIM * DIM * 2;
    _Float16* qd    = (_Float16*)ws; ws += (size_t)MTOT * DIM * 2;
    _Float16* kd    = (_Float16*)ws; ws += (size_t)MTOT * DIM * 2;
    _Float16* vd    = (_Float16*)ws; ws += (size_t)MTOT * DIM * 2;
    _Float16* ctxh  = (_Float16*)ws; ws += (size_t)MTOT * DIM * 2;

    cvt_f32_to_f16_kernel<<<(MTOT * DIM) / 256, 256, 0, stream>>>(x, xh, MTOT * DIM);
    transpose_cvt_kernel<<<(DIM * NQKV) / 256, 256, 0, stream>>>(w_qkv, wqkvT, DIM, NQKV);
    transpose_cvt_kernel<<<(DIM * DIM) / 256, 256, 0, stream>>>(w_out, woutT, DIM, DIM);

    qkv_gemm_kernel<<<dim3(NQKV / 128, MTOT / 128), 256, 0, stream>>>(xh, wqkvT, qd, kd, vd);
    attn_kernel<<<dim3(SEQ / 128, BATCH * HEADS), 256, 0, stream>>>(qd, kd, vd, ctxh);
    out_gemm_kernel<<<dim3(DIM / 128, MTOT / 128), 256, 0, stream>>>(ctxh, woutT, out);
}
